// Gemma3TextModel_3934190043259
// MI455X (gfx1250) — matrix operand order repaired and checked by reading, not compiled, not run on hardware
//
#include <hip/hip_runtime.h>
#include <hip/hip_bf16.h>
#include <cstdint>
#include <cstddef>

// ---------------- model constants ----------------
#define BB   2
#define SS   1024
#define HHD  2560
#define NH   32
#define NKV  8
#define DHD  128
#define GQ   4            // NH / NKV
#define IID  10240
#define MM   (BB*SS)      // 2048 tokens
#define WIN  256

__device__ __constant__ float kScaling = 0.1889822365046136f;  // 28^-0.5
__device__ __constant__ float kSoftcap = 50.0f;

// ---------------- WMMA types ----------------
typedef __attribute__((ext_vector_type(16))) __bf16 v16bf;
typedef __attribute__((ext_vector_type(8)))  float  v8f;

union Frag16 {
  v16bf v;
  uint4 q[2];
  __bf16 b[16];
};

__device__ __forceinline__ v8f wmma_bf16(const Frag16& a, const Frag16& b, v8f c) {
  return __builtin_amdgcn_wmma_f32_16x16x32_bf16(
      /*neg_a=*/false, a.v, /*neg_b=*/false, b.v,
      /*c_mod=*/(short)0, c, /*reuse_a=*/false, /*reuse_b=*/false);
}

// ---------------- shared reduction helper ----------------
__device__ __forceinline__ float row_rms_inv(const float* __restrict__ xr, int N,
                                             float* __restrict__ sp) {
  float ss = 0.f;
  for (int c = threadIdx.x; c < N; c += blockDim.x) { float t = xr[c]; ss += t * t; }
  for (int off = 16; off; off >>= 1) ss += __shfl_xor(ss, off, 32);
  int lane = threadIdx.x & 31, wv = threadIdx.x >> 5;
  if (lane == 0) sp[wv] = ss;
  __syncthreads();
  if (threadIdx.x == 0) {
    float t = 0.f;
    int nw = blockDim.x >> 5;
    for (int i = 0; i < nw; ++i) t += sp[i];
    sp[0] = rsqrtf(t / (float)N + 1e-6f);
  }
  __syncthreads();
  return sp[0];
}

// ---------------- embedding gather + sqrt(H) scale ----------------
__global__ __launch_bounds__(256)
void k_embed(const int* __restrict__ ids, const float* __restrict__ emb,
             float* __restrict__ h) {
  int i = blockIdx.x * blockDim.x + threadIdx.x;      // group of 4 floats
  int total = MM * HHD / 4;
  if (i >= total) return;
  int idx = i * 4;
  int m = idx / HHD;
  int c = idx - m * HHD;
  float sc = sqrtf((float)HHD);
  float4 e = *(const float4*)(emb + (size_t)ids[m] * HHD + c);
  float4 r; r.x = e.x * sc; r.y = e.y * sc; r.z = e.z * sc; r.w = e.w * sc;
  *(float4*)(h + (size_t)m * HHD + c) = r;
}

// ---------------- RMSNorm -> bf16 activations ----------------
__global__ __launch_bounds__(256)
void k_rmsnorm_bf16(const float* __restrict__ x, const float* __restrict__ w,
                    __bf16* __restrict__ out, int N) {
  __shared__ float sp[8];
  int row = blockIdx.x;
  const float* xr = x + (size_t)row * N;
  float inv = row_rms_inv(xr, N, sp);
  __bf16* orow = out + (size_t)row * N;
  for (int c = threadIdx.x; c < N; c += blockDim.x)
    orow[c] = (__bf16)(xr[c] * inv * (1.f + w[c]));
}

// ---------------- h += RMSNorm(x) ----------------
__global__ __launch_bounds__(256)
void k_resid_rmsnorm(float* __restrict__ h, const float* __restrict__ x,
                     const float* __restrict__ w, int N) {
  __shared__ float sp[8];
  int row = blockIdx.x;
  const float* xr = x + (size_t)row * N;
  float inv = row_rms_inv(xr, N, sp);
  float* hrow = h + (size_t)row * N;
  for (int c = threadIdx.x; c < N; c += blockDim.x)
    hrow[c] += xr[c] * inv * (1.f + w[c]);
}

// ---------------- final RMSNorm -> fp32 out ----------------
__global__ __launch_bounds__(256)
void k_rmsnorm_f32(const float* __restrict__ x, const float* __restrict__ w,
                   float* __restrict__ out, int N) {
  __shared__ float sp[8];
  int row = blockIdx.x;
  const float* xr = x + (size_t)row * N;
  float inv = row_rms_inv(xr, N, sp);
  float* orow = out + (size_t)row * N;
  for (int c = threadIdx.x; c < N; c += blockDim.x)
    orow[c] = xr[c] * inv * (1.f + w[c]);
}

// ---------------- GEMM: C[M,N] = A_bf16[M,K] * B_f32[K,N] (f32 accum/out) ----
// Block tile 128x128, K-step 32. 8 waves in 4(M) x 2(N) grid; each wave computes
// a 32x64 strip = 2 A-frags x 4 B-frags = 8 WMMAs per K-step.
__global__ __launch_bounds__(256)
void k_gemm(const __bf16* __restrict__ A, const float* __restrict__ Bw,
            float* __restrict__ C, int N, int K) {
  __shared__ __bf16 sA[128][32];    // row-major A tile   (8 KB)
  __shared__ __bf16 sBt[128][32];   // B tile transposed: [n][k] (8 KB)

  const int tid  = threadIdx.x;
  const int wave = tid >> 5;
  const int lane = tid & 31;
  const int half = lane >> 4;
  const int l15  = lane & 15;
  const int m0   = blockIdx.y * 128;
  const int n0   = blockIdx.x * 128;
  const int msub = (wave & 3) * 32;
  const int nsub = (wave >> 2) * 64;

  v8f acc[2][4];
  #pragma unroll
  for (int mi = 0; mi < 2; ++mi)
    #pragma unroll
    for (int ni = 0; ni < 4; ++ni)
      #pragma unroll
      for (int i = 0; i < 8; ++i) acc[mi][ni][i] = 0.f;

  const int arow = tid >> 1, acg = (tid & 1) * 16;   // A loader: 16 bf16 each
  const int brow = tid >> 3, bcg = (tid & 7) * 16;   // B loader: 16 f32 each

  for (int kk0 = 0; kk0 < K; kk0 += 32) {
    // --- stage A tile (128x32 bf16) ---
    const __bf16* ag = A + (size_t)(m0 + arow) * K + kk0 + acg;
    *(uint4*)&sA[arow][acg]     = *(const uint4*)ag;
    *(uint4*)&sA[arow][acg + 8] = *(const uint4*)(ag + 8);

    // --- stage B tile (32x128 f32 -> bf16, transposed to [n][k]) ---
    const float* bg = Bw + (size_t)(kk0 + brow) * N + n0 + bcg;
    #pragma unroll
    for (int v4 = 0; v4 < 4; ++v4) {
      float4 f = *(const float4*)(bg + v4 * 4);
      sBt[bcg + v4 * 4 + 0][brow] = (__bf16)f.x;
      sBt[bcg + v4 * 4 + 1][brow] = (__bf16)f.y;
      sBt[bcg + v4 * 4 + 2][brow] = (__bf16)f.z;
      sBt[bcg + v4 * 4 + 3][brow] = (__bf16)f.w;
    }

    if (kk0 + 32 < K) {                      // gfx1250 global_prefetch_b8
      __builtin_prefetch((const void*)(ag + 32), 0, 0);
      __builtin_prefetch((const void*)(bg + (size_t)32 * N), 0, 0);
    }
    __syncthreads();

    // A fragments (16x32 bf16): row = msub + mi*16 + l15; k runs [8*half..] and [16+8*half..]
    Frag16 a[2];
    #pragma unroll
    for (int mi = 0; mi < 2; ++mi) {
      a[mi].q[0] = *(uint4*)&sA[msub + mi * 16 + l15][8 * half];
      a[mi].q[1] = *(uint4*)&sA[msub + mi * 16 + l15][16 + 8 * half];
    }
    // B fragments (32x16 bf16): col = nsub + ni*16 + l15; k runs [8*half..] and [16+8*half..]
    Frag16 bfr[4];
    #pragma unroll
    for (int ni = 0; ni < 4; ++ni) {
      bfr[ni].q[0] = *(uint4*)&sBt[nsub + ni * 16 + l15][8 * half];
      bfr[ni].q[1] = *(uint4*)&sBt[nsub + ni * 16 + l15][16 + 8 * half];
    }

    #pragma unroll
    for (int mi = 0; mi < 2; ++mi)
      #pragma unroll
      for (int ni = 0; ni < 4; ++ni)
        acc[mi][ni] = wmma_bf16(a[mi], bfr[ni], acc[mi][ni]);
    __syncthreads();
  }

  #pragma unroll
  for (int mi = 0; mi < 2; ++mi)
    #pragma unroll
    for (int ni = 0; ni < 4; ++ni)
      #pragma unroll
      for (int r = 0; r < 8; ++r) {
        int mr = m0 + msub + mi * 16 + r + half * 8;
        C[(size_t)mr * N + n0 + nsub + ni * 16 + l15] = acc[mi][ni][r];
      }
}

// ---------------- per-head q/k RMSNorm + RoPE (or plain bf16 repack for v) ----
// grid: M * nheads blocks of 128 threads; out layout [B, nheads, S, DH] bf16
__global__ __launch_bounds__(128)
void k_head_prep(const float* __restrict__ x, const float* __restrict__ w,
                 __bf16* __restrict__ out, int nheads, float rope_base, int do_nr) {
  __shared__ float sh[DHD];
  __shared__ float red[4];
  int bi = blockIdx.x;
  int m  = bi / nheads;
  int hh = bi - m * nheads;
  int d  = threadIdx.x;
  int b  = m / SS;
  int s  = m - b * SS;
  float v = x[(size_t)m * nheads * DHD + (size_t)hh * DHD + d];
  float y = v;
  if (do_nr) {
    float ss = v * v;
    for (int off = 16; off; off >>= 1) ss += __shfl_xor(ss, off, 32);
    int lane = d & 31, wv = d >> 5;
    if (lane == 0) red[wv] = ss;
    __syncthreads();
    float tot = red[0] + red[1] + red[2] + red[3];
    float inv = rsqrtf(tot / (float)DHD + 1e-6f);
    y = v * inv * (1.f + w[d]);
    sh[d] = y;
    __syncthreads();
    int   d2   = d & 63;
    float invf = __powf(rope_base, -(float)(2 * d2) / (float)DHD);
    float f    = (float)s * invf;
    float c = cosf(f), sn = sinf(f);
    float rot = (d < 64) ? -sh[d + 64] : sh[d - 64];
    y = y * c + rot * sn;
  }
  out[(((size_t)b * nheads + hh) * SS + s) * DHD + d] = (__bf16)y;
}

// ---------------- flash attention, one wave per (b, head, 16-query tile) ------
__global__ __launch_bounds__(32)
void k_attn(const __bf16* __restrict__ Qb, const __bf16* __restrict__ Kb,
            const __bf16* __restrict__ Vb, __bf16* __restrict__ Ob, int sliding) {
  __shared__ float  sS[16][32];
  __shared__ __bf16 sP[16][32];
  __shared__ float  sAlpha[16];
  __shared__ float  sL[16];
  __shared__ __bf16 sV[32][DHD];

  const int qt = blockIdx.x, hh = blockIdx.y, b = blockIdx.z;
  const int kv = hh / GQ;
  const int lane = threadIdx.x;
  const int half = lane >> 4;
  const int l15  = lane & 15;
  const int q0   = qt * 16;
  const int row  = l15;
  const int qi   = q0 + row;

  const __bf16* Qh = Qb + ((size_t)(b * NH  + hh) * SS) * DHD;
  const __bf16* Kh = Kb + ((size_t)(b * NKV + kv) * SS) * DHD;
  const __bf16* Vh = Vb + ((size_t)(b * NKV + kv) * SS) * DHD;

  // Q fragments: 4 k-steps of 32 over DH=128
  Frag16 qa[4];
  {
    const __bf16* qp = Qh + (size_t)(q0 + l15) * DHD;
    #pragma unroll
    for (int kk = 0; kk < 4; ++kk) {
      int base = kk * 32 + half * 8;
      qa[kk].q[0] = *(const uint4*)(qp + base);
      qa[kk].q[1] = *(const uint4*)(qp + base + 16);
    }
  }

  v8f acc[8];
  #pragma unroll
  for (int t = 0; t < 8; ++t)
    #pragma unroll
    for (int i = 0; i < 8; ++i) acc[t][i] = 0.f;

  float mrow = -1e30f, lrow = 0.f;
  const int npair = (q0 + 15) / 32 + 1;   // causal: keys <= q0+15

  for (int kp = 0; kp < npair; ++kp) {
    const int kbase = kp * 32;

    // stage V pair-tile (32 keys x 128) into LDS, vectorized by key row
    {
      const __bf16* vp = Vh + (size_t)(kbase + lane) * DHD;
      #pragma unroll
      for (int j = 0; j < 16; ++j)
        *(uint4*)&sV[lane][j * 8] = *(const uint4*)(vp + j * 8);
    }

    // scores for two 16-key tiles
    #pragma unroll
    for (int sub = 0; sub < 2; ++sub) {
      int k0 = kbase + sub * 16;
      v8f sacc;
      #pragma unroll
      for (int i = 0; i < 8; ++i) sacc[i] = 0.f;
      #pragma unroll
      for (int kk = 0; kk < 4; ++kk) {
        Frag16 kb;
        const __bf16* kptr = Kh + (size_t)(k0 + l15) * DHD + kk * 32 + half * 8;
        kb.q[0] = *(const uint4*)kptr;
        kb.q[1] = *(const uint4*)(kptr + 16);
        sacc = wmma_bf16(qa[kk], kb, sacc);
      }
      #pragma unroll
      for (int r = 0; r < 8; ++r)
        sS[r + half * 8][sub * 16 + l15] = sacc[r];
    }
    __syncthreads();

    // online softmax with tanh softcap (lanes L and L+16 redundantly do row L)
    float xs[32];
    float tmax = -1e30f;
    #pragma unroll
    for (int c = 0; c < 32; ++c) {
      int   j  = kbase + c;
      float sc = sS[row][c];
      float xx = tanhf(sc * (kScaling / kSoftcap)) * kSoftcap;
      bool  msk = (j > qi) || (sliding && (j >= qi + WIN));
      xs[c] = msk ? -1e30f : xx;
      tmax  = fmaxf(tmax, xs[c]);
    }
    float mnew  = fmaxf(mrow, tmax);
    float alpha = __expf(mrow - mnew);
    float lsum  = 0.f;
    #pragma unroll
    for (int c = 0; c < 32; ++c) {
      float p = __expf(xs[c] - mnew);
      lsum += p;
      sP[row][c] = (__bf16)p;
    }
    lrow = lrow * alpha + lsum;
    mrow = mnew;
    sAlpha[row] = alpha;
    __syncthreads();

    // rescale accumulators per-row
    #pragma unroll
    for (int r = 0; r < 8; ++r) {
      float a = sAlpha[r + half * 8];
      #pragma unroll
      for (int t = 0; t < 8; ++t) acc[t][r] *= a;
    }

    // P (16x32) x V (32x128): 8 WMMAs
    Frag16 pa;
    pa.q[0] = *(uint4*)&sP[row][half * 16];
    pa.q[1] = *(uint4*)&sP[row][half * 16 + 8];
    #pragma unroll
    for (int t = 0; t < 8; ++t) {
      Frag16 vb;
      int ncol = t * 16 + l15;
      #pragma unroll
      for (int i = 0; i < 16; ++i)
        vb.b[i] = sV[i + 16 * half][ncol];
      acc[t] = wmma_bf16(pa, vb, acc[t]);
    }
    __syncthreads();
  }

  sL[row] = lrow;
  __syncthreads();

  #pragma unroll
  for (int r = 0; r < 8; ++r) {
    float invl = 1.f / sL[r + half * 8];
    int   mg   = q0 + r + half * 8;
    size_t obase = ((size_t)b * SS + mg) * (size_t)(NH * DHD) + (size_t)hh * DHD;
    #pragma unroll
    for (int t = 0; t < 8; ++t)
      Ob[obase + t * 16 + l15] = (__bf16)(acc[t][r] * invl);
  }
}

// ---------------- silu(gate)*up -> bf16 ----------------
__global__ __launch_bounds__(256)
void k_silu_mul(const float* __restrict__ g, const float* __restrict__ u,
                __bf16* __restrict__ act, long long n) {
  long long i = (long long)blockIdx.x * blockDim.x + threadIdx.x;
  if (i >= n) return;
  float gv = g[i];
  float s  = gv / (1.f + __expf(-gv));
  act[i]   = (__bf16)(s * u[i]);
}

// ---------------- host launcher ----------------
extern "C" void kernel_launch(void* const* d_in, const int* in_sizes, int n_in,
                              void* d_out, int out_size, void* d_ws, size_t ws_size,
                              hipStream_t stream) {
  const int*   ids      = (const int*)d_in[0];
  const float* embed    = (const float*)d_in[1];
  const float* Wq       = (const float*)d_in[2];
  const float* Wk       = (const float*)d_in[3];
  const float* Wv       = (const float*)d_in[4];
  const float* Wo       = (const float*)d_in[5];
  const float* q_norm_w = (const float*)d_in[6];
  const float* k_norm_w = (const float*)d_in[7];
  const float* in_ln    = (const float*)d_in[8];
  const float* post_a   = (const float*)d_in[9];
  const float* pre_f    = (const float*)d_in[10];
  const float* post_f   = (const float*)d_in[11];
  const float* Wgate    = (const float*)d_in[12];
  const float* Wup      = (const float*)d_in[13];
  const float* Wdown    = (const float*)d_in[14];
  const float* fin_w    = (const float*)d_in[15];
  float* out            = (float*)d_out;

  char* p = (char*)d_ws;
  auto alloc = [&](size_t bytes) -> void* {
    void* r = (void*)p;
    p += (bytes + 255) & ~(size_t)255;
    return r;
  };
  float*  h    = (float*)alloc((size_t)MM * HHD * 4);
  __bf16* hn   = (__bf16*)alloc((size_t)MM * HHD * 2);
  float*  qf   = (float*)alloc((size_t)MM * NH * DHD * 4);
  float*  kf   = (float*)alloc((size_t)MM * NKV * DHD * 4);
  float*  vf   = (float*)alloc((size_t)MM * NKV * DHD * 4);
  __bf16* Qb   = (__bf16*)alloc((size_t)MM * NH * DHD * 2);
  __bf16* Kb   = (__bf16*)alloc((size_t)MM * NKV * DHD * 2);
  __bf16* Vb   = (__bf16*)alloc((size_t)MM * NKV * DHD * 2);
  __bf16* Ob   = (__bf16*)alloc((size_t)MM * NH * DHD * 2);
  float*  proj = (float*)alloc((size_t)MM * HHD * 4);
  float*  gf   = (float*)alloc((size_t)MM * IID * 4);
  float*  uf   = (float*)alloc((size_t)MM * IID * 4);
  __bf16* act  = (__bf16*)alloc((size_t)MM * IID * 2);
  (void)ws_size; (void)n_in; (void)in_sizes; (void)out_size;

  const float rope_base[2] = {10000.f, 500000.f};
  const int   slide[2]     = {0, 1};

  // embedding
  {
    int total4 = MM * HHD / 4;
    k_embed<<<(total4 + 255) / 256, 256, 0, stream>>>(ids, embed, h);
  }

  for (int l = 0; l < 2; ++l) {
    const float* wq = Wq + (size_t)l * HHD * (NH * DHD);
    const float* wk = Wk + (size_t)l * HHD * (NKV * DHD);
    const float* wv = Wv + (size_t)l * HHD * (NKV * DHD);
    const float* wo = Wo + (size_t)l * (NH * DHD) * HHD;
    const float* wg = Wgate + (size_t)l * HHD * IID;
    const float* wu = Wup   + (size_t)l * HHD * IID;
    const float* wd = Wdown + (size_t)l * IID * HHD;

    // pre-attn norm
    k_rmsnorm_bf16<<<MM, 256, 0, stream>>>(h, in_ln + (size_t)l * HHD, hn, HHD);

    // qkv projections
    k_gemm<<<dim3((NH * DHD) / 128, MM / 128), 256, 0, stream>>>(hn, wq, qf, NH * DHD, HHD);
    k_gemm<<<dim3((NKV * DHD) / 128, MM / 128), 256, 0, stream>>>(hn, wk, kf, NKV * DHD, HHD);
    k_gemm<<<dim3((NKV * DHD) / 128, MM / 128), 256, 0, stream>>>(hn, wv, vf, NKV * DHD, HHD);

    // per-head norm + rope, repack to [B, heads, S, DH] bf16
    k_head_prep<<<MM * NH,  128, 0, stream>>>(qf, q_norm_w + (size_t)l * DHD, Qb, NH,  rope_base[l], 1);
    k_head_prep<<<MM * NKV, 128, 0, stream>>>(kf, k_norm_w + (size_t)l * DHD, Kb, NKV, rope_base[l], 1);
    k_head_prep<<<MM * NKV, 128, 0, stream>>>(vf, nullptr,                    Vb, NKV, 0.f,          0);

    // attention
    k_attn<<<dim3(SS / 16, NH, BB), 32, 0, stream>>>(Qb, Kb, Vb, Ob, slide[l]);

    // output projection + residual
    k_gemm<<<dim3(HHD / 128, MM / 128), 256, 0, stream>>>(Ob, wo, proj, HHD, NH * DHD);
    k_resid_rmsnorm<<<MM, 256, 0, stream>>>(h, proj, post_a + (size_t)l * HHD, HHD);

    // MLP
    k_rmsnorm_bf16<<<MM, 256, 0, stream>>>(h, pre_f + (size_t)l * HHD, hn, HHD);
    k_gemm<<<dim3(IID / 128, MM / 128), 256, 0, stream>>>(hn, wg, gf, IID, HHD);
    k_gemm<<<dim3(IID / 128, MM / 128), 256, 0, stream>>>(hn, wu, uf, IID, HHD);
    {
      long long n = (long long)MM * IID;
      k_silu_mul<<<(unsigned)((n + 255) / 256), 256, 0, stream>>>(gf, uf, act, n);
    }
    k_gemm<<<dim3(HHD / 128, MM / 128), 256, 0, stream>>>(act, wd, proj, HHD, IID);
    k_resid_rmsnorm<<<MM, 256, 0, stream>>>(h, proj, post_f + (size_t)l * HHD, HHD);
  }

  // final norm -> fp32 output
  k_rmsnorm_f32<<<MM, 256, 0, stream>>>(h, fin_w, out, HHD);
}
